// EncoderLayer_82617990905987
// MI455X (gfx1250) — matrix operand order repaired and checked by reading, not compiled, not run on hardware
//
#include <hip/hip_runtime.h>
#include <math.h>

// ---------------------------------------------------------------------------
// Types for CDNA5 WMMA (wave32)
// ---------------------------------------------------------------------------
typedef __bf16 bf16;
typedef __bf16 v16bf __attribute__((ext_vector_type(16)));
typedef __bf16 v8bf  __attribute__((ext_vector_type(8)));
typedef float  v8f   __attribute__((ext_vector_type(8)));

struct alignas(16) U4 { unsigned int x, y, z, w; };

union BfOp { v16bf v; U4 u[2]; unsigned int w[8]; };  // WMMA operand, 32B/lane
union Bf8  { v8bf  v; U4 u;    };                     // 16-byte bf16 store
union BfPk { __bf16 h[2]; unsigned int u; };          // packed bf16 pair

static __device__ __forceinline__ v8f wmma_bf16(v16bf a, v16bf b, v8f c) {
  // D(16x16 f32) = A(16x32 bf16) * B(32x16 bf16) + C
  return __builtin_amdgcn_wmma_f32_16x16x32_bf16(false, a, false, b,
                                                 (short)0, c, false, false);
}

// Problem constants (fixed by the reference)
constexpr int B_  = 4;
constexpr int S_  = 2048;
constexpr int E_  = 1024;
constexpr int H_  = 16;
constexpr int D_  = 64;
constexpr int F_  = 4096;
constexpr int TOK = B_ * S_;   // 8192 tokens

// ---------------------------------------------------------------------------
// fp32 -> bf16 convert (weights)
// ---------------------------------------------------------------------------
__global__ __launch_bounds__(256) void cvt_kernel(const float* __restrict__ in,
                                                  bf16* __restrict__ out, int n) {
  int i = blockIdx.x * 256 + threadIdx.x;
  if (i < n) out[i] = (bf16)in[i];
}

// ---------------------------------------------------------------------------
// LayerNorm (unbiased std, eps added to std) -> bf16 output
// One block (256 threads) per row of E_=1024.
// ---------------------------------------------------------------------------
__global__ __launch_bounds__(256) void ln_kernel(const float* __restrict__ x,
                                                 const float* __restrict__ w,
                                                 const float* __restrict__ b,
                                                 bf16* __restrict__ out) {
  const int row = blockIdx.x;
  const float* xr = x + (size_t)row * E_;
  float v[4];
  float s = 0.f, ss = 0.f;
#pragma unroll
  for (int i = 0; i < 4; ++i) {
    v[i] = xr[threadIdx.x + i * 256];
    s  += v[i];
    ss += v[i] * v[i];
  }
#pragma unroll
  for (int off = 16; off > 0; off >>= 1) {
    s  += __shfl_xor(s,  off, 32);
    ss += __shfl_xor(ss, off, 32);
  }
  __shared__ float rs[8], rss[8];
  const int wid = threadIdx.x >> 5, lane = threadIdx.x & 31;
  if (lane == 0) { rs[wid] = s; rss[wid] = ss; }
  __syncthreads();
  s = 0.f; ss = 0.f;
#pragma unroll
  for (int i = 0; i < 8; ++i) { s += rs[i]; ss += rss[i]; }
  const float mean = s / (float)E_;
  float var = (ss - s * mean) / (float)(E_ - 1);
  var = fmaxf(var, 0.f);
  const float inv = 1.f / (sqrtf(var) + 1e-6f);
#pragma unroll
  for (int i = 0; i < 4; ++i) {
    const int j = threadIdx.x + i * 256;
    out[(size_t)row * E_ + j] = (bf16)(w[j] * (v[i] - mean) * inv + b[j]);
  }
}

// ---------------------------------------------------------------------------
// WMMA GEMM: C[M,N] = A[M,K] @ W[N,K]^T  (A, W bf16 row-major; fp32 acc)
// Block tile 256x128, 8 waves (4x2), wave tile 64x64, K-step 32, LDS staged.
// 16 WMMAs per wave per K-step vs 16 ds_load_b128 -> high WMMA/LDS ratio.
// MODE 0: QKV scatter (q scaled by 1/sqrt(D); v stored transposed [B,H,D,S])
// MODE 1: out = resid + val + bias      -> fp32
// MODE 2: out = gelu(val + bias)        -> bf16
// MODE 3: out = resid + val + bias      -> fp32 (final output)
// ---------------------------------------------------------------------------
template <int MODE>
__global__ __launch_bounds__(256) void gemm_kernel(
    const bf16* __restrict__ A, const bf16* __restrict__ W,
    const float* __restrict__ bias, const float* __restrict__ resid,
    float* __restrict__ outF, bf16* __restrict__ outB,
    bf16* __restrict__ qo, bf16* __restrict__ ko, bf16* __restrict__ vto,
    int M, int N, int Kdim) {
  __shared__ bf16 As[256][40];
  __shared__ bf16 Bs[128][40];

  const int tid  = threadIdx.x;
  const int wid  = tid >> 5;
  const int lane = tid & 31;
  const int lh   = lane >> 4;     // half-wave id (0/1)
  const int l16  = lane & 15;
  const int wm   = wid >> 1;      // 0..3
  const int wn   = wid & 1;       // 0..1
  const int m0   = blockIdx.y * 256;
  const int n0   = blockIdx.x * 128;
  const int brow = tid >> 1;      // B staging row 0..127
  const int bhal = tid & 1;       // B staging half (16 elems)

  v8f acc[4][4] = {};

  const bf16* gA = A + (size_t)(m0 + tid) * Kdim;          // one full row
  const bf16* gB = W + (size_t)(n0 + brow) * Kdim + bhal * 16;

  for (int kt = 0; kt < Kdim; kt += 32) {
    // stage A: 256 rows x 32 cols (64B per thread = whole row)
    const U4* pa = (const U4*)(gA + kt);
    U4 a0 = pa[0], a1 = pa[1], a2 = pa[2], a3 = pa[3];
    // stage B: 128 rows x 32 cols (32B per thread = half row)
    const U4* pb = (const U4*)(gB + kt);
    U4 b0 = pb[0], b1 = pb[1];
    if (kt + 32 < Kdim) {   // prefetch next K tile -> global_prefetch_b8
      __builtin_prefetch((const void*)(gA + kt + 32), 0, 3);
      __builtin_prefetch((const void*)(gB + kt + 32), 0, 3);
    }
    __syncthreads();
    *(U4*)&As[tid][0]  = a0;
    *(U4*)&As[tid][8]  = a1;
    *(U4*)&As[tid][16] = a2;
    *(U4*)&As[tid][24] = a3;
    *(U4*)&Bs[brow][bhal * 16]     = b0;
    *(U4*)&Bs[brow][bhal * 16 + 8] = b1;
    __syncthreads();

    // B operands: lane holds K = lh*8+0..7 and 16+lh*8+0..7 for column n = l16
    BfOp bop[4];
#pragma unroll
    for (int ns = 0; ns < 4; ++ns) {
      const int n = wn * 64 + ns * 16 + l16;
      bop[ns].u[0] = *(const U4*)&Bs[n][lh * 8];
      bop[ns].u[1] = *(const U4*)&Bs[n][16 + lh * 8];
    }
#pragma unroll
    for (int ms = 0; ms < 4; ++ms) {
      // A operand: lane holds K = lh*8+0..7 and 16+lh*8+0..7 for row m
      const int m = wm * 64 + ms * 16 + l16;
      BfOp aop;
      aop.u[0] = *(const U4*)&As[m][lh * 8];
      aop.u[1] = *(const U4*)&As[m][16 + lh * 8];
#pragma unroll
      for (int ns = 0; ns < 4; ++ns)
        acc[ms][ns] = wmma_bf16(aop.v, bop[ns].v, acc[ms][ns]);
    }
  }

  // epilogue: C layout -> row = i + lh*8 (per VGPR i), col = l16
#pragma unroll
  for (int ms = 0; ms < 4; ++ms) {
#pragma unroll
    for (int ns = 0; ns < 4; ++ns) {
      const int gn = n0 + wn * 64 + ns * 16 + l16;
#pragma unroll
      for (int i = 0; i < 8; ++i) {
        const int gm = m0 + wm * 64 + ms * 16 + i + lh * 8;
        const float val = acc[ms][ns][i];
        if (MODE == 0) {
          const int chunk = gn >> 10, e = gn & 1023;
          const int hh = e >> 6, dd = e & 63;
          const int bb = gm >> 11, s = gm & 2047;
          if (chunk == 0)
            qo[(((size_t)bb * H_ + hh) * S_ + s) * D_ + dd] = (bf16)(val * 0.125f);
          else if (chunk == 1)
            ko[(((size_t)bb * H_ + hh) * S_ + s) * D_ + dd] = (bf16)val;
          else
            vto[(((size_t)bb * H_ + hh) * D_ + dd) * S_ + s] = (bf16)val;
        } else if (MODE == 1 || MODE == 3) {
          outF[(size_t)gm * N + gn] = resid[(size_t)gm * N + gn] + val + bias[gn];
        } else {  // MODE == 2 : exact GELU
          const float t = val + bias[gn];
          const float g = 0.5f * t * (1.f + erff(t * 0.70710678118654752f));
          outB[(size_t)gm * N + gn] = (bf16)g;
        }
      }
    }
  }
}

// ---------------------------------------------------------------------------
// Flash attention: per wave, 16 query rows; S^T = K*Q^T (rows=keys, cols=q)
// so online softmax stats are lane-local; O^T = V^T * P^T keeps rescale
// per-lane. V pre-transposed to [B,H,D,S]. Q pre-scaled by 1/sqrt(D).
// Half-wave P exchange done on PACKED bf16 pairs (8 shuffles, not 16).
// Grid: (S/64, B*H), 128 threads (4 waves).
// ---------------------------------------------------------------------------
__global__ __launch_bounds__(128) void attn_kernel(const bf16* __restrict__ Q,
                                                   const bf16* __restrict__ K,
                                                   const bf16* __restrict__ Vt,
                                                   bf16* __restrict__ ctx) {
  const int bh   = blockIdx.y;          // 0..63
  const int bb   = bh >> 4;
  const int hh   = bh & 15;
  const int wid  = threadIdx.x >> 5;
  const int lane = threadIdx.x & 31;
  const int lh   = lane >> 4;
  const int l16  = lane & 15;
  const int qrow = blockIdx.x * 64 + wid * 16 + l16;

  // Q as B operand (Q^T): lane = column q, K = d chunks (lh*8, 16+lh*8) (+32)
  const bf16* qb = Q + ((size_t)bh * S_ + qrow) * D_;
  BfOp qb0, qb1;
  qb0.u[0] = *(const U4*)(qb + lh * 8);
  qb0.u[1] = *(const U4*)(qb + 16 + lh * 8);
  qb1.u[0] = *(const U4*)(qb + 32 + lh * 8);
  qb1.u[1] = *(const U4*)(qb + 48 + lh * 8);

  v8f o[4] = {};
  float m = -1e30f, l = 0.f;

  for (int j = 0; j < S_; j += 32) {
    // K tiles as A operands: row = key, K = d chunks (lh*8, 16+lh*8) (+32)
    const bf16* k0 = K + ((size_t)bh * S_ + j + l16) * D_;
    const bf16* k1 = k0 + 16 * D_;
    BfOp a0lo, a0hi, a1lo, a1hi;
    a0lo.u[0] = *(const U4*)(k0 + lh * 8);
    a0lo.u[1] = *(const U4*)(k0 + 16 + lh * 8);
    a0hi.u[0] = *(const U4*)(k0 + 32 + lh * 8);
    a0hi.u[1] = *(const U4*)(k0 + 48 + lh * 8);
    a1lo.u[0] = *(const U4*)(k1 + lh * 8);
    a1lo.u[1] = *(const U4*)(k1 + 16 + lh * 8);
    a1hi.u[0] = *(const U4*)(k1 + 32 + lh * 8);
    a1hi.u[1] = *(const U4*)(k1 + 48 + lh * 8);

    v8f st0 = {}, st1 = {};
    st0 = wmma_bf16(a0lo.v, qb0.v, st0);   // keys j..j+15, d 0..31
    st0 = wmma_bf16(a0hi.v, qb1.v, st0);   //               d 32..63
    st1 = wmma_bf16(a1lo.v, qb0.v, st1);   // keys j+16..j+31
    st1 = wmma_bf16(a1hi.v, qb1.v, st1);

    // lane owns q = l16; holds keys {i+lh*8} (st0) and {16+i+lh*8} (st1)
    float mx = -1e30f;
#pragma unroll
    for (int i = 0; i < 8; ++i) mx = fmaxf(mx, fmaxf(st0[i], st1[i]));
    mx = fmaxf(mx, __shfl_xor(mx, 16, 32));
    const float mnew  = fmaxf(m, mx);
    const float alpha = __expf(m - mnew);

    float p0[8], p1[8];
    float sum = 0.f;
#pragma unroll
    for (int i = 0; i < 8; ++i) {
      p0[i] = __expf(st0[i] - mnew);
      p1[i] = __expf(st1[i] - mnew);
      sum += p0[i] + p1[i];
    }
    sum += __shfl_xor(sum, 16, 32);
    l = l * alpha + sum;
    m = mnew;
#pragma unroll
    for (int t = 0; t < 4; ++t)
#pragma unroll
      for (int i = 0; i < 8; ++i) o[t][i] *= alpha;

    // Pack P into bf16 pairs FIRST, then exchange packed words (8 shuffles).
    unsigned int P0[4], P1[4];
#pragma unroll
    for (int v = 0; v < 4; ++v) {
      BfPk pk;
      pk.h[0] = (bf16)p0[2 * v];
      pk.h[1] = (bf16)p0[2 * v + 1];
      P0[v] = pk.u;
      pk.h[0] = (bf16)p1[2 * v];
      pk.h[1] = (bf16)p1[2 * v + 1];
      P1[v] = pk.u;
    }
    unsigned int X0[4], X1[4];
#pragma unroll
    for (int v = 0; v < 4; ++v) {
      X0[v] = (unsigned int)__shfl_xor((int)P0[v], 16, 32);
      X1[v] = (unsigned int)__shfl_xor((int)P1[v], 16, 32);
    }
    // B operand P^T: lane needs keys lh*16 .. lh*16+15 for column q = l16.
    // lh=0: keys 0..7 own (P0), keys 8..15 from partner (X0);
    // lh=1: keys 16..23 from partner (X1), keys 24..31 own (P1).
    BfOp pbop;
#pragma unroll
    for (int v = 0; v < 4; ++v) {
      pbop.w[v]     = lh ? X1[v] : P0[v];
      pbop.w[4 + v] = lh ? P1[v] : X0[v];
    }

    // O^T += V^T * P^T : A = Vt tile (rows d, K = keys), B = pbop
#pragma unroll
    for (int t = 0; t < 4; ++t) {
      const bf16* vb = Vt + ((size_t)bh * D_ + t * 16 + l16) * S_ + j;
      BfOp va;
      va.u[0] = *(const U4*)(vb + lh * 16);
      va.u[1] = *(const U4*)(vb + lh * 16 + 8);
      o[t] = wmma_bf16(va.v, pbop.v, o[t]);
    }
  }

  // write ctx[token, hh*64 + d]; lane holds q=l16, rows d = t*16 + i + lh*8
  const float inv = 1.f / l;
  const size_t token = (size_t)bb * S_ + qrow;
#pragma unroll
  for (int t = 0; t < 4; ++t) {
    Bf8 w8;
#pragma unroll
    for (int i = 0; i < 8; ++i) w8.v[i] = (bf16)(o[t][i] * inv);
    *(U4*)(ctx + token * E_ + hh * D_ + t * 16 + lh * 8) = w8.u;
  }
}

// ---------------------------------------------------------------------------
// Host-side launcher
// ---------------------------------------------------------------------------
extern "C" void kernel_launch(void* const* d_in, const int* in_sizes, int n_in,
                              void* d_out, int out_size, void* d_ws, size_t ws_size,
                              hipStream_t stream) {
  (void)in_sizes; (void)n_in; (void)out_size; (void)ws_size;

  const float* x     = (const float*)d_in[0];
  const float* w_qkv = (const float*)d_in[1];
  const float* w_out = (const float*)d_in[2];
  const float* b_out = (const float*)d_in[3];
  const float* w_fc1 = (const float*)d_in[4];
  const float* b_fc1 = (const float*)d_in[5];
  const float* w_fc2 = (const float*)d_in[6];
  const float* b_fc2 = (const float*)d_in[7];
  const float* ln1w  = (const float*)d_in[8];
  const float* ln1b  = (const float*)d_in[9];
  const float* ln2w  = (const float*)d_in[10];
  const float* ln2b  = (const float*)d_in[11];

  char* ws = (char*)d_ws;
  size_t off = 0;
  auto take = [&](size_t bytes) { void* p = ws + off; off += bytes; return p; };

  bf16*  xn   = (bf16*)take((size_t)TOK * E_ * 2);
  bf16*  qbuf = (bf16*)take((size_t)TOK * E_ * 2);
  bf16*  kbuf = (bf16*)take((size_t)TOK * E_ * 2);
  bf16*  vtb  = (bf16*)take((size_t)TOK * E_ * 2);
  bf16*  ctx  = (bf16*)take((size_t)TOK * E_ * 2);
  float* x1   = (float*)take((size_t)TOK * E_ * 4);
  bf16*  xn2  = (bf16*)take((size_t)TOK * E_ * 2);
  bf16*  hbuf = (bf16*)take((size_t)TOK * F_ * 2);
  bf16*  wqb  = (bf16*)take((size_t)3 * E_ * E_ * 2);
  bf16*  wob  = (bf16*)take((size_t)E_ * E_ * 2);
  bf16*  w1b  = (bf16*)take((size_t)F_ * E_ * 2);
  bf16*  w2b  = (bf16*)take((size_t)E_ * F_ * 2);

  // 1) weights -> bf16
  cvt_kernel<<<(3 * E_ * E_) / 256, 256, 0, stream>>>(w_qkv, wqb, 3 * E_ * E_);
  cvt_kernel<<<(E_ * E_) / 256, 256, 0, stream>>>(w_out, wob, E_ * E_);
  cvt_kernel<<<(F_ * E_) / 256, 256, 0, stream>>>(w_fc1, w1b, F_ * E_);
  cvt_kernel<<<(E_ * F_) / 256, 256, 0, stream>>>(w_fc2, w2b, E_ * F_);

  // 2) ln1
  ln_kernel<<<TOK, 256, 0, stream>>>(x, ln1w, ln1b, xn);

  // 3) QKV projection (scatter into q [scaled], k, v-transposed)
  gemm_kernel<0><<<dim3(3 * E_ / 128, TOK / 256), 256, 0, stream>>>(
      xn, wqb, nullptr, nullptr, nullptr, nullptr, qbuf, kbuf, vtb,
      TOK, 3 * E_, E_);

  // 4) attention -> ctx (bf16)
  attn_kernel<<<dim3(S_ / 64, B_ * H_), 128, 0, stream>>>(qbuf, kbuf, vtb, ctx);

  // 5) out projection + residual -> x1 (fp32)
  gemm_kernel<1><<<dim3(E_ / 128, TOK / 256), 256, 0, stream>>>(
      ctx, wob, b_out, x, x1, nullptr, nullptr, nullptr, nullptr,
      TOK, E_, E_);

  // 6) ln2
  ln_kernel<<<TOK, 256, 0, stream>>>(x1, ln2w, ln2b, xn2);

  // 7) FC1 + GELU -> h (bf16)
  gemm_kernel<2><<<dim3(F_ / 128, TOK / 256), 256, 0, stream>>>(
      xn2, w1b, b_fc1, nullptr, nullptr, hbuf, nullptr, nullptr, nullptr,
      TOK, F_, E_);

  // 8) FC2 + residual -> output (fp32)
  gemm_kernel<3><<<dim3(E_ / 128, TOK / 256), 256, 0, stream>>>(
      hbuf, w2b, b_fc2, x1, (float*)d_out, nullptr, nullptr, nullptr, nullptr,
      TOK, E_, F_);
}
